// TriangleMultiplication_49744311222639
// MI455X (gfx1250) — compile-verified, matrix operand order checked by reading, not run on hardware
//
#include <hip/hip_runtime.h>
#include <cstdint>
#include <cstddef>

#define NN 768
#define CC 128
#define NPOS (NN * NN)
#define PLANE ((size_t)NN * (size_t)NN)

typedef __attribute__((ext_vector_type(16))) __bf16 v16bf;
typedef __attribute__((ext_vector_type(8)))  float  v8f;

// ---------- helpers ----------
static __device__ __forceinline__ unsigned short f2bfbits(float f) {
  unsigned u = __builtin_bit_cast(unsigned, f);
  unsigned r = u + 0x7FFFu + ((u >> 16) & 1u);   // round-to-nearest-even
  return (unsigned short)(r >> 16);
}
static __device__ __forceinline__ __bf16 f2bf(float f) {
  unsigned short s = f2bfbits(f);
  return __builtin_bit_cast(__bf16, s);
}
static __device__ __forceinline__ float sigf(float x) {
  return 1.0f / (1.0f + __expf(-x));
}
static __device__ __forceinline__ uint4 pack8(const float* v) {
  uint4 q;
  q.x = (unsigned)f2bfbits(v[0]) | ((unsigned)f2bfbits(v[1]) << 16);
  q.y = (unsigned)f2bfbits(v[2]) | ((unsigned)f2bfbits(v[3]) << 16);
  q.z = (unsigned)f2bfbits(v[4]) | ((unsigned)f2bfbits(v[5]) << 16);
  q.w = (unsigned)f2bfbits(v[6]) | ((unsigned)f2bfbits(v[7]) << 16);
  return q;
}
// pre-swizzled weight fragment: one contiguous 32B load per lane
static __device__ __forceinline__ v16bf load_wfrag(const __bf16* W, int nt, int kt, int lane) {
  return *reinterpret_cast<const v16bf*>(W + (((size_t)(nt * 4 + kt)) * 32 + lane) * 16);
}
// gather a 16xK=32 WMMA fragment from an LDS row-major strip.
// ISA 7.12.2 (16-bit A 16x32): lanes 0-15 -> K {0..7,16..23}; lanes 16-31 -> K {8..15,24..31}
static __device__ __forceinline__ v16bf lds_frag(const __bf16* base, int row, int stride,
                                                 int kbase, int hi) {
  v16bf f;
  const __bf16* p = base + row * stride + kbase + hi * 8;
#pragma unroll
  for (int e = 0; e < 8; ++e) f[e] = p[e];
#pragma unroll
  for (int e = 0; e < 8; ++e) f[8 + e] = p[16 + e];
  return f;
}
static __device__ __forceinline__ v8f wmma_bf16(v16bf a, v16bf b, v8f c) {
  return __builtin_amdgcn_wmma_f32_16x16x32_bf16(false, a, false, b, (short)0, c, false, false);
}

// ---------- CDNA5 async global->LDS copy (ASYNCcnt DMA), guarded fallback ----------
static __device__ __forceinline__ void cp_async16(const void* g, void* l) {
#if __has_builtin(__builtin_amdgcn_global_load_async_to_lds_b128)
  typedef int vs4i __attribute__((vector_size(16)));
  __builtin_amdgcn_global_load_async_to_lds_b128((vs4i*)g, (vs4i*)l, 0, 0);
#else
  *reinterpret_cast<uint4*>(l) = *reinterpret_cast<const uint4*>(g);
#endif
}
static __device__ __forceinline__ void wait_async_all() {
#if __has_builtin(__builtin_amdgcn_s_wait_asynccnt)
  __builtin_amdgcn_s_wait_asynccnt(0);
#elif defined(__gfx1250__)
  asm volatile("s_wait_asynccnt 0" ::: "memory");
#endif
}
static __device__ __forceinline__ void nt_store(float* p, float v) {
  __builtin_nontemporal_store(v, p);
}

// ---------- kernel 0: convert + swizzle weights to WMMA B-fragment order ----------
// WIN n-tiles: [0..15]=w_g_in(256 cols) [16..31]=w_p_in(256) [32..39]=w_g_out(128)
__global__ void prep_weights(const float* __restrict__ wg_in, const float* __restrict__ wp_in,
                             const float* __restrict__ wg_out, const float* __restrict__ wp_out,
                             __bf16* __restrict__ WIN, __bf16* __restrict__ WPOUT) {
  int fg = blockIdx.x;            // 0..191
  int nt = fg >> 2, kt = fg & 3;  // 48 n-tiles x 4 k-tiles (K=128)
  int lane = threadIdx.x;         // wave32
  const float* src; int nbase, ld; __bf16* dst; size_t fo;
  if (nt < 16)      { src = wg_in;  nbase = nt * 16;        ld = 256; dst = WIN;
                      fo = (((size_t)(nt * 4 + kt)) * 32 + lane) * 16; }
  else if (nt < 32) { src = wp_in;  nbase = (nt - 16) * 16; ld = 256; dst = WIN;
                      fo = (((size_t)(nt * 4 + kt)) * 32 + lane) * 16; }
  else if (nt < 40) { src = wg_out; nbase = (nt - 32) * 16; ld = 128; dst = WIN;
                      fo = (((size_t)(nt * 4 + kt)) * 32 + lane) * 16; }
  else              { src = wp_out; nbase = (nt - 40) * 16; ld = 128; dst = WPOUT;
                      fo = (((size_t)((nt - 40) * 4 + kt)) * 32 + lane) * 16; }
  int n = nbase + (lane & 15);
  int hi = lane >> 4;
#pragma unroll
  for (int e = 0; e < 16; ++e) {
    int k = kt * 32 + hi * 8 + (e & 7) + ((e >> 3) << 4);
    dst[fo + e] = f2bf(src[(size_t)k * ld + n]);
  }
}

// ---------- kernel 1: LN + (xn@Wg)*sig gating + g_out, 16 positions/block ----------
__global__ void __launch_bounds__(256) fused_in(
    const float* __restrict__ x, const float* __restrict__ mask,
    const float* __restrict__ lnw, const float* __restrict__ lnb,
    const __bf16* __restrict__ WIN,
    __bf16* __restrict__ a_t, __bf16* __restrict__ b_t, float* __restrict__ g_out) {
  __shared__ __align__(16) __bf16 xnS[16 * 136];   // bf16 xn tile, padded stride
  __shared__ float gst[16 * 132];                  // g_out staging (transpose)
  __shared__ float maskS[16];
  int tid = threadIdx.x;
  int p0 = blockIdx.x * 16;
  int i = p0 / NN, j0 = p0 % NN;

  // LayerNorm over C: 16 lanes x 8 channels per position, shfl reduce (wave32, width 16)
  int pos = tid >> 4, l16 = tid & 15;
  const float* xr = x + (size_t)(p0 + pos) * CC + l16 * 8;
  float v[8];
  float4 f0 = *reinterpret_cast<const float4*>(xr);
  float4 f1 = *reinterpret_cast<const float4*>(xr + 4);
  v[0]=f0.x; v[1]=f0.y; v[2]=f0.z; v[3]=f0.w; v[4]=f1.x; v[5]=f1.y; v[6]=f1.z; v[7]=f1.w;
  float s = 0.f, sq = 0.f;
#pragma unroll
  for (int e = 0; e < 8; ++e) { s += v[e]; sq += v[e] * v[e]; }
  for (int o = 8; o >= 1; o >>= 1) { s += __shfl_xor(s, o, 16); sq += __shfl_xor(sq, o, 16); }
  float mean = s * (1.0f / CC);
  float rstd = rsqrtf(sq * (1.0f / CC) - mean * mean + 1e-5f);
#pragma unroll
  for (int e = 0; e < 8; ++e) {
    int c = l16 * 8 + e;
    xnS[pos * 136 + c] = f2bf((v[e] - mean) * rstd * lnw[c] + lnb[c]);
  }
  if (l16 == 0) maskS[pos] = mask[p0 + pos];
  __syncthreads();

  // WMMA: wave w owns n-tiles {w, w+8, w+16, w+24, w+32} -> pairs (G,P) stay in-wave
  int w = tid >> 5, lane = tid & 31, nl = lane & 15, hi = lane >> 4;
  v8f aGa = {}, aGb = {}, aPa = {}, aPb = {}, aO = {};
#pragma unroll
  for (int kt = 0; kt < 4; ++kt) {
    v16bf af  = lds_frag(xnS, nl, 136, kt * 32, hi);
    v16bf bGa = load_wfrag(WIN, w,      kt, lane);
    v16bf bGb = load_wfrag(WIN, w + 8,  kt, lane);
    v16bf bPa = load_wfrag(WIN, w + 16, kt, lane);
    v16bf bPb = load_wfrag(WIN, w + 24, kt, lane);
    v16bf bO  = load_wfrag(WIN, w + 32, kt, lane);
    aGa = wmma_bf16(af, bGa, aGa);
    aGb = wmma_bf16(af, bGb, aGb);
    aPa = wmma_bf16(af, bPa, aPa);
    aPb = wmma_bf16(af, bPb, aPb);
    aO  = wmma_bf16(af, bO,  aO);
  }
  // gating; D layout: lane=N col, VGPR r -> row r (lanes 0-15) / r+8 (lanes 16-31)
  float av[8], bv[8];
#pragma unroll
  for (int r = 0; r < 8; ++r) {
    float mv = maskS[r + hi * 8];
    av[r] = sigf(aGa[r]) * aPa[r] * mv;
    bv[r] = sigf(aGb[r]) * aPb[r];
    gst[(r + hi * 8) * 132 + w * 16 + nl] = sigf(aO[r]);
  }
  int c = w * 16 + nl;  // channel (0..127) for both a and b
  size_t rowoff = (size_t)c * PLANE + (size_t)i * NN + (size_t)(j0 + hi * 8);
  *reinterpret_cast<uint4*>(a_t + rowoff) = pack8(av);   // channel-major, contiguous 16B
  *reinterpret_cast<uint4*>(b_t + rowoff) = pack8(bv);
  __syncthreads();
  for (int idx = tid; idx < 16 * 128; idx += 256) {      // coalesced g_out write
    int pp = idx >> 7, ch = idx & 127;
    g_out[(size_t)(p0 + pp) * CC + ch] = gst[pp * 132 + ch];
  }
}

// ---------- kernel 2: tri[c] = A_c * B_c^T ----------
// 128x128 block tile per (c, i0, j0); K-panels of 32, double-buffered LDS filled by
// async global->LDS DMA; each wave: 2 i-subtiles x 4 j-subtiles = 8 WMMAs / panel.
#define TMROWS 128
#define TMSTRIDE 40
#define TMBUF (TMROWS * TMSTRIDE)
__global__ void __launch_bounds__(256) tri_mm(const __bf16* __restrict__ a_t,
                                              const __bf16* __restrict__ b_t,
                                              float* __restrict__ tri_t) {
  __shared__ __align__(16) __bf16 As[2 * TMBUF];
  __shared__ __align__(16) __bf16 Bs[2 * TMBUF];
  int tid = threadIdx.x;
  int c = blockIdx.y;
  int it = blockIdx.x / 6, jt = blockIdx.x % 6;
  int i0 = it * 128, j0 = jt * 128;
  int w = tid >> 5, lane = tid & 31, nl = lane & 15, hi = lane >> 4;
  int iw = w & 3;    // 2 i-subtiles: iw*2, iw*2+1
  int jw = w >> 2;   // 4 j-subtiles: jw*4 .. jw*4+3
  v8f acc[2][4] = {};

  const __bf16* Abase = a_t + (size_t)c * PLANE + (size_t)i0 * NN;
  const __bf16* Bbase = b_t + (size_t)c * PLANE + (size_t)j0 * NN;

  auto copy_panel = [&](int kc, int buf) {
    int k0 = kc * 32;
    __bf16* Ad = As + buf * TMBUF;
    __bf16* Bd = Bs + buf * TMBUF;
#pragma unroll
    for (int t = 0; t < 4; ++t) {
      int q = tid + t * 256;            // 0..1023 : 512 A chunks + 512 B chunks (16B each)
      int isB = q >> 9;
      int row = (q & 511) >> 2;
      int seg = (q & 3) * 8;
      const __bf16* g = (isB ? Bbase : Abase) + (size_t)row * NN + k0 + seg;
      __bf16* l = (isB ? Bd : Ad) + row * TMSTRIDE + seg;
      cp_async16(g, l);
    }
  };

  copy_panel(0, 0);
  wait_async_all();
  __syncthreads();
  for (int kc = 0; kc < 24; ++kc) {
    int buf = kc & 1;
    if (kc + 1 < 24) copy_panel(kc + 1, buf ^ 1);   // DMA overlaps WMMAs below
    const __bf16* Asb = As + buf * TMBUF;
    const __bf16* Bsb = Bs + buf * TMBUF;
    v16bf af0 = lds_frag(Asb, (iw * 2 + 0) * 16 + nl, TMSTRIDE, 0, hi);
    v16bf af1 = lds_frag(Asb, (iw * 2 + 1) * 16 + nl, TMSTRIDE, 0, hi);
#pragma unroll
    for (int jj = 0; jj < 4; ++jj) {
      v16bf bfr = lds_frag(Bsb, (jw * 4 + jj) * 16 + nl, TMSTRIDE, 0, hi);
      acc[0][jj] = wmma_bf16(af0, bfr, acc[0][jj]);
      acc[1][jj] = wmma_bf16(af1, bfr, acc[1][jj]);
    }
    wait_async_all();
    __syncthreads();
  }
#pragma unroll
  for (int ii = 0; ii < 2; ++ii) {
#pragma unroll
    for (int jj = 0; jj < 4; ++jj) {
      int j = j0 + (jw * 4 + jj) * 16 + nl;
      int ib = i0 + (iw * 2 + ii) * 16 + hi * 8;
#pragma unroll
      for (int r = 0; r < 8; ++r)
        tri_t[(size_t)c * PLANE + (size_t)(ib + r) * NN + j] = acc[ii][jj][r];
    }
  }
}

// ---------- kernel 3: LN(tri) @ w_p_out * g_out ----------
__global__ void __launch_bounds__(256) fused_out(
    const float* __restrict__ tri_t, const float* __restrict__ g_out,
    const float* __restrict__ lnw, const float* __restrict__ lnb,
    const __bf16* __restrict__ WPOUT, float* __restrict__ out) {
  __shared__ float trs[16 * 129];                 // tri tile; reused as out staging
  __shared__ float gs[16 * 132];
  __shared__ __align__(16) __bf16 xnS[16 * 136];
  int tid = threadIdx.x;
  int p0 = blockIdx.x * 16;
  int i = p0 / NN, j0 = p0 % NN;
  for (int idx = tid; idx < 2048; idx += 256) {   // gather channel-major tri (64B/16 lanes)
    int ch = idx >> 4, pp = idx & 15;
    trs[pp * 129 + ch] = tri_t[(size_t)ch * PLANE + (size_t)i * NN + (j0 + pp)];
  }
  for (int idx = tid; idx < 2048; idx += 256) {   // coalesced g_out load
    int pp = idx >> 7, ch = idx & 127;
    gs[pp * 132 + ch] = g_out[(size_t)(p0 + pp) * CC + ch];
  }
  __syncthreads();
  // LayerNorm over channels
  int pos = tid >> 4, l16 = tid & 15;
  float v[8];
#pragma unroll
  for (int e = 0; e < 8; ++e) v[e] = trs[pos * 129 + l16 * 8 + e];
  float s = 0.f, sq = 0.f;
#pragma unroll
  for (int e = 0; e < 8; ++e) { s += v[e]; sq += v[e] * v[e]; }
  for (int o = 8; o >= 1; o >>= 1) { s += __shfl_xor(s, o, 16); sq += __shfl_xor(sq, o, 16); }
  float mean = s * (1.0f / CC);
  float rstd = rsqrtf(sq * (1.0f / CC) - mean * mean + 1e-5f);
#pragma unroll
  for (int e = 0; e < 8; ++e) {
    int c = l16 * 8 + e;
    xnS[pos * 136 + c] = f2bf((v[e] - mean) * rstd * lnw[c] + lnb[c]);
  }
  __syncthreads();
  // projection + gate
  int w = tid >> 5, lane = tid & 31, nl = lane & 15, hi = lane >> 4;
  v8f acc = {};
#pragma unroll
  for (int kt = 0; kt < 4; ++kt) {
    v16bf af = lds_frag(xnS, nl, 136, kt * 32, hi);
    v16bf bf = load_wfrag(WPOUT, w, kt, lane);
    acc = wmma_bf16(af, bf, acc);
  }
#pragma unroll
  for (int r = 0; r < 8; ++r)
    trs[(r + hi * 8) * 129 + w * 16 + nl] = acc[r] * gs[(r + hi * 8) * 132 + w * 16 + nl];
  __syncthreads();
  for (int idx = tid; idx < 2048; idx += 256) {   // coalesced NT fp32 output
    int pp = idx >> 7, ch = idx & 127;
    nt_store(out + (size_t)(p0 + pp) * CC + ch, trs[pp * 129 + ch]);
  }
}

// ---------- launch ----------
extern "C" void kernel_launch(void* const* d_in, const int* in_sizes, int n_in,
                              void* d_out, int out_size, void* d_ws, size_t ws_size,
                              hipStream_t stream) {
  (void)in_sizes; (void)n_in; (void)out_size; (void)ws_size;
  const float* x       = (const float*)d_in[0];
  const float* mask    = (const float*)d_in[1];
  const float* nin_w   = (const float*)d_in[2];
  const float* nin_b   = (const float*)d_in[3];
  const float* w_g_in  = (const float*)d_in[4];
  const float* w_p_in  = (const float*)d_in[5];
  const float* nout_w  = (const float*)d_in[6];
  const float* nout_b  = (const float*)d_in[7];
  const float* w_g_out = (const float*)d_in[8];
  const float* w_p_out = (const float*)d_in[9];
  float* out = (float*)d_out;

  char* ws = (char*)d_ws;
  const size_t AB_BYTES = (size_t)CC * PLANE * 2;   // bf16 plane stack: 151 MB
  const size_t F_BYTES  = (size_t)CC * PLANE * 4;   // fp32: 302 MB
  __bf16* WIN   = (__bf16*)(ws);                          // 160 KB (L2-resident)
  __bf16* WPOUT = (__bf16*)(ws + 163840);                 // 32 KB
  __bf16* a_t   = (__bf16*)(ws + 196608);                 // [c][i][k] bf16
  __bf16* b_t   = (__bf16*)(ws + 196608 + AB_BYTES);      // [c][j][k] bf16
  float*  gbuf  = (float*) (ws + 196608 + 2 * AB_BYTES);  // [pos][c] f32
  float*  tri_t = (float*) (ws + 196608 + 2 * AB_BYTES + F_BYTES);  // [c][i][j] f32

  prep_weights<<<192, 32, 0, stream>>>(w_g_in, w_p_in, w_g_out, w_p_out, WIN, WPOUT);
  fused_in<<<NPOS / 16, 256, 0, stream>>>(x, mask, nin_w, nin_b, WIN, a_t, b_t, gbuf);
  tri_mm<<<dim3(36, 128), 256, 0, stream>>>(a_t, b_t, tri_t);
  fused_out<<<NPOS / 16, 256, 0, stream>>>(tri_t, gbuf, nout_w, nout_b, WPOUT, out);
}
